// HMP_TFNModel_77017353552141
// MI455X (gfx1250) — matrix operand order repaired and checked by reading, not compiled, not run on hardware
//
#include <hip/hip_runtime.h>
#include <hip/hip_bf16.h>
#include <math.h>

// ---------------------------------------------------------------------------
// Problem constants (match reference)
// ---------------------------------------------------------------------------
#define NN   8192     // nodes
#define EE   65536    // edges
#define NG_  8
#define EMB_ 32
#define SH_  9
#define HID_ 288
#define SDIM_ 16
#define MM   2048     // selected nodes
#define KV_  16
#define MKV  (MM*KV_)        // 32768
#define MKV2 (2*MM*KV_)      // 65536
#define ME_  (EE + MKV2)     // 131072
#define NB_  8
#define MLPD_ 64
#define MSH_ 32
#define NL_  3

typedef __attribute__((ext_vector_type(16))) _Float16 v16h;
typedef __attribute__((ext_vector_type(8)))  float    v8f;

union V16HU { v16h v; unsigned u[8]; _Float16 h[16]; };
union H2U   { _Float16 h[2]; unsigned u; };

__device__ __forceinline__ unsigned encf(float f) {
    unsigned u = __float_as_uint(f);
    return (u & 0x80000000u) ? ~u : (u | 0x80000000u);
}
__device__ __forceinline__ float decf(unsigned e) {
    return __uint_as_float((e & 0x80000000u) ? (e ^ 0x80000000u) : ~e);
}

// ---------------------------------------------------------------------------
// Generic fills
// ---------------------------------------------------------------------------
__global__ void fill_f32(float* p, int n, float v) {
    int i = blockIdx.x * blockDim.x + threadIdx.x;
    if (i < n) p[i] = v;
}
__global__ void fill_u32(unsigned* p, int n, unsigned v) {
    int i = blockIdx.x * blockDim.x + threadIdx.x;
    if (i < n) p[i] = v;
}

// ---------------------------------------------------------------------------
// Embedding: hA[n, :32] = emb[atoms[n]], rest zero
// ---------------------------------------------------------------------------
__global__ void embed_kernel(const int* __restrict__ atoms,
                             const float* __restrict__ emb,
                             float* __restrict__ hA, int total) {
    int idx = blockIdx.x * blockDim.x + threadIdx.x;
    if (idx >= total) return;
    int n = idx / HID_, c = idx - n * HID_;
    hA[idx] = (c < EMB_) ? emb[atoms[n] * EMB_ + c] : 0.f;
}

// ---------------------------------------------------------------------------
// Pack W (in x 9 x 288) into WMMA B-fragment layout, f16:
// Wp[((ctG*nchunks + c)*8 + j)*32 + lane] = half2( W'[c*32+kr][n], W'[..+1][n] )
// where kr = 2j + 16*(lane>=16), n = ctG*16 + (lane&15), W'[s*in+k][o]=W[k][s][o]
// ---------------------------------------------------------------------------
__global__ void wpack_kernel(const float* __restrict__ W, int Kin, int nchunks,
                             unsigned* __restrict__ Wp) {
    int idx = blockIdx.x * blockDim.x + threadIdx.x;
    int total = 18 * nchunks * 8 * 32;
    if (idx >= total) return;
    int lane = idx & 31;
    int j    = (idx >> 5) & 7;
    int c    = (idx >> 8) % nchunks;
    int ctG  = (idx >> 8) / nchunks;
    int kr = 2 * j + 16 * (lane >> 4);
    int n  = ctG * 16 + (lane & 15);
    int kappa = c * 32 + kr;
    int s = kappa / Kin;
    int k = kappa - s * Kin;
    H2U h2;
    h2.h[0] = (_Float16)W[((size_t)k * SH_ + s) * HID_ + n];
    h2.h[1] = (_Float16)W[((size_t)(k + 1) * SH_ + s) * HID_ + n];
    Wp[idx] = h2.u;
}

// ---------------------------------------------------------------------------
// Per-edge spherical harmonics * radial-MLP coefficients (9 per edge)
// Optional per-edge ef scale (wmul) and msg validity (validv) folded in.
// ---------------------------------------------------------------------------
__global__ __launch_bounds__(256)
void coeff_kernel(const float* __restrict__ pos,
                  const int* __restrict__ src, const int* __restrict__ dst, int nedges,
                  const float* __restrict__ rw1, const float* __restrict__ rb1,
                  const float* __restrict__ rw2, const float* __restrict__ rb2,
                  const float* __restrict__ wmul, const float* __restrict__ validv,
                  float* __restrict__ coeff) {
    __shared__ float s_rw1[NB_ * MLPD_];
    __shared__ float s_rb1[MLPD_];
    __shared__ float s_rw2[MLPD_ * SH_];
    __shared__ float s_rb2[SH_];
    for (int t = threadIdx.x; t < NB_ * MLPD_; t += blockDim.x) s_rw1[t] = rw1[t];
    for (int t = threadIdx.x; t < MLPD_; t += blockDim.x) s_rb1[t] = rb1[t];
    for (int t = threadIdx.x; t < MLPD_ * SH_; t += blockDim.x) s_rw2[t] = rw2[t];
    if (threadIdx.x < SH_) s_rb2[threadIdx.x] = rb2[threadIdx.x];
    __syncthreads();
    int e = blockIdx.x * blockDim.x + threadIdx.x;
    if (e >= nedges) return;
    int s0 = src[e], d0 = dst[e];
    float dx = pos[s0 * 3 + 0] - pos[d0 * 3 + 0];
    float dy = pos[s0 * 3 + 1] - pos[d0 * 3 + 1];
    float dz = pos[s0 * 3 + 2] - pos[d0 * 3 + 2];
    float r  = sqrtf(dx * dx + dy * dy + dz * dz);
    float rc = fmaxf(r, 1e-9f);
    float ux = dx / rc, uy = dy / rc, uz = dz / rc;
    const float s3 = 1.7320508075688772f, s5 = 2.2360679774997896f, s15 = 3.872983346207417f;
    float Y[9];
    Y[0] = 1.f;            Y[1] = s3 * uy;        Y[2] = s3 * uz;
    Y[3] = s3 * ux;        Y[4] = s15 * ux * uy;  Y[5] = s15 * uy * uz;
    Y[6] = 0.5f * s5 * (3.f * uz * uz - 1.f);
    Y[7] = s15 * ux * uz;  Y[8] = 0.5f * s15 * (ux * ux - uy * uy);
    float x = r * 0.1f;
    float x2 = x * x;
    float x5 = x2 * x2 * x;
    float env = 1.f - 21.f * x5 + 35.f * x5 * x - 15.f * x5 * x2;
    if (x >= 1.f) env = 0.f;
    float amp = 0.4472135954999579f * env / rc;   // sqrt(2/RMAX)
    float ef[NB_];
#pragma unroll
    for (int nb = 0; nb < NB_; ++nb)
        ef[nb] = amp * sinf((float)(nb + 1) * 3.14159265358979323846f * x);
    if (wmul) {
        float w = wmul[e];
#pragma unroll
        for (int nb = 0; nb < NB_; ++nb) ef[nb] *= w;
    }
    float acc9[SH_];
#pragma unroll
    for (int s = 0; s < SH_; ++s) acc9[s] = s_rb2[s];
    for (int o = 0; o < MLPD_; ++o) {
        float z = s_rb1[o];
#pragma unroll
        for (int b = 0; b < NB_; ++b) z += ef[b] * s_rw1[b * MLPD_ + o];
        z = fmaxf(z, 0.f);
#pragma unroll
        for (int s = 0; s < SH_; ++s) acc9[s] += z * s_rw2[o * SH_ + s];
    }
    float vv = validv ? validv[e] : 1.f;
#pragma unroll
    for (int s = 0; s < SH_; ++s) coeff[(size_t)e * SH_ + s] = Y[s] * acc9[s] * vv;
}

// ---------------------------------------------------------------------------
// WMMA tensor-product edge conv:  out[dst_e, :] += A' @ W'
// Block: 128 threads (4 waves), 64 edges x 96 output cols; gridDim.y = 3
// Double-buffered A tile in LDS: gather/convert of chunk c+1 overlaps the
// WMMAs of chunk c; one barrier per K-chunk.
// ---------------------------------------------------------------------------
__global__ __launch_bounds__(128)
void conv_wmma_kernel(const float* __restrict__ h, int Kin,
                      const int* __restrict__ src, const int* __restrict__ dst,
                      const float* __restrict__ coeff, int nedges,
                      const unsigned* __restrict__ Wpack, int nchunks,
                      float* __restrict__ out) {
    __shared__ alignas(16) _Float16 A_lds[2][64][36];
    __shared__ float coeff_s[64][10];
    __shared__ int src_s[64];
    __shared__ int dst_s[64];
    const int tid  = threadIdx.x;
    const int lane = tid & 31;
    const int rt   = tid >> 5;            // wave id = row tile
    const int ebase = blockIdx.x * 64;
    const int cg    = blockIdx.y;         // col group of 96

    if (tid < 64) {
        int eg = ebase + tid;
        int sv = 0, dv = 0;
        if (eg < nedges) { sv = src[eg]; dv = dst[eg]; }
        src_s[tid] = sv; dst_s[tid] = dv;
    }
    for (int t = tid; t < 64 * SH_; t += 128) {
        int e = t / SH_, s = t - e * SH_;
        int eg = ebase + e;
        coeff_s[e][s] = (eg < nedges) ? coeff[(size_t)eg * SH_ + s] : 0.f;
    }
    __syncthreads();

    v8f acc[6] = {};
    const int e_loc = tid & 63;
    const int gsel  = tid >> 6;           // 0/1 -> halves of the 32-wide K chunk
    const float* hbase = h + (size_t)src_s[e_loc] * HID_;
    unsigned* aw0 = (unsigned*)&A_lds[0][e_loc][gsel * 16];
    unsigned* aw1p = (unsigned*)&A_lds[1][e_loc][gsel * 16];

    auto build = [&](int c, int b) {
        int sIdx  = (c * 32) / Kin;
        int kbase = c * 32 - sIdx * Kin;
        float cs = coeff_s[e_loc][sIdx];
        const float* hp = hbase + kbase + gsel * 16;
        unsigned* wout = b ? aw1p : aw0;
#pragma unroll
        for (int u = 0; u < 8; ++u) {
            H2U p;
            p.h[0] = (_Float16)(cs * hp[2 * u]);
            p.h[1] = (_Float16)(cs * hp[2 * u + 1]);
            wout[u] = p.u;
        }
    };

    build(0, 0);
    __syncthreads();

    int buf = 0;
    for (int c = 0; c < nchunks; ++c) {
        if (c + 1 < nchunks) {
            // prefetch next B-fragment block while gathering next A chunk
            __builtin_prefetch(Wpack + ((size_t)((cg * 6) * nchunks + c + 1) * 8) * 32 + lane, 0, 1);
            build(c + 1, buf ^ 1);
        }
        V16HU a;
        const unsigned* arow = (const unsigned*)&A_lds[buf][rt * 16 + (lane & 15)][0];
#pragma unroll
        for (int j = 0; j < 8; ++j) {
            int k = 2 * j + 16 * (lane >> 4);
            a.u[j] = arow[k >> 1];
        }
#pragma unroll
        for (int ct = 0; ct < 6; ++ct) {
            V16HU b;
            const unsigned* bp = Wpack + ((size_t)((cg * 6 + ct) * nchunks + c) * 8) * 32 + lane;
#pragma unroll
            for (int j = 0; j < 8; ++j) b.u[j] = bp[j * 32];
            acc[ct] = __builtin_amdgcn_wmma_f32_16x16x32_f16(
                false, a.v, false, b.v, (short)0, acc[ct], false, false);
        }
        __syncthreads();
        buf ^= 1;
    }
#pragma unroll
    for (int ct = 0; ct < 6; ++ct) {
#pragma unroll
        for (int j = 0; j < 8; ++j) {
            int row = rt * 16 + j + 8 * (lane >> 4);
            int eg = ebase + row;
            if (eg < nedges) {
                int col = cg * 96 + ct * 16 + (lane & 15);
                atomicAdd(&out[(size_t)dst_s[row] * HID_ + col], acc[ct][j]);
            }
        }
    }
}

// ---------------------------------------------------------------------------
// h_local = h_up + pad(h_prev)
// ---------------------------------------------------------------------------
__global__ void addpad_kernel(const float* __restrict__ hA, float* __restrict__ hB,
                              int wprev, int N_) {
    int idx = blockIdx.x * blockDim.x + threadIdx.x;
    if (idx >= N_ * wprev) return;
    int n = idx / wprev, c = idx - n * wprev;
    hB[(size_t)n * HID_ + c] += hA[(size_t)n * HID_ + c];
}

// ---------------------------------------------------------------------------
// node gate m = sigmoid(MLP(h_local[:, :16]))
// ---------------------------------------------------------------------------
__global__ __launch_bounds__(256)
void mgate_kernel(const float* __restrict__ hB,
                  const float* __restrict__ w1, const float* __restrict__ b1,
                  const float* __restrict__ w2, const float* __restrict__ b2,
                  float* __restrict__ m, int N_) {
    __shared__ float s_w1[SDIM_ * MSH_];
    __shared__ float s_b1[MSH_];
    __shared__ float s_w2[MSH_];
    __shared__ float s_b2;
    for (int t = threadIdx.x; t < SDIM_ * MSH_; t += blockDim.x) s_w1[t] = w1[t];
    if (threadIdx.x < MSH_) { s_b1[threadIdx.x] = b1[threadIdx.x]; s_w2[threadIdx.x] = w2[threadIdx.x]; }
    if (threadIdx.x == 0) s_b2 = b2[0];
    __syncthreads();
    int n = blockIdx.x * blockDim.x + threadIdx.x;
    if (n >= N_) return;
    float hs[SDIM_];
#pragma unroll
    for (int k = 0; k < SDIM_; ++k) hs[k] = hB[(size_t)n * HID_ + k];
    float z2 = s_b2;
    for (int o = 0; o < MSH_; ++o) {
        float z = s_b1[o];
#pragma unroll
        for (int k = 0; k < SDIM_; ++k) z += hs[k] * s_w1[k * MSH_ + o];
        z2 += fmaxf(z, 0.f) * s_w2[o];
    }
    m[n] = 1.f / (1.f + expf(-z2));
}

// ---------------------------------------------------------------------------
// Radix-select the value of the Msel-th largest gate (keys are positive floats)
// ---------------------------------------------------------------------------
__global__ __launch_bounds__(256)
void thresh_kernel(const float* __restrict__ m, int n, int Msel, unsigned* __restrict__ outT) {
    __shared__ int sh[256];
    unsigned prefix = 0;
    int remaining = Msel;
    for (int bit = 31; bit >= 0; --bit) {
        unsigned candHi = (prefix >> bit) | 1u;
        int c = 0;
        for (int i = threadIdx.x; i < n; i += 256) {
            unsigned u = __float_as_uint(m[i]);
            if ((u >> bit) == candHi) c++;
        }
        sh[threadIdx.x] = c;
        __syncthreads();
        for (int off = 128; off > 0; off >>= 1) {
            if (threadIdx.x < off) sh[threadIdx.x] += sh[threadIdx.x + off];
            __syncthreads();
        }
        int total = sh[0];
        __syncthreads();
        if (total >= remaining) prefix |= (1u << bit);
        else remaining -= total;
    }
    if (threadIdx.x == 0) outT[0] = prefix;
}

// ---------------------------------------------------------------------------
// Build sorted midx (ascending node index, jax tie order) and inverse map
// ---------------------------------------------------------------------------
__global__ __launch_bounds__(256)
void select_kernel(const float* __restrict__ m, int n, int Msel,
                   const unsigned* __restrict__ Tptr,
                   int* __restrict__ midx, int* __restrict__ inv) {
    __shared__ int s_gt[257], s_eq[257];
    unsigned T = Tptr[0];
    int per = n / 256;
    int start = threadIdx.x * per;
    int gt = 0, eq = 0;
    for (int i = 0; i < per; ++i) {
        unsigned u = __float_as_uint(m[start + i]);
        if (u > T) gt++; else if (u == T) eq++;
    }
    s_gt[threadIdx.x] = gt; s_eq[threadIdx.x] = eq;
    __syncthreads();
    if (threadIdx.x == 0) {
        int ag = 0, ae = 0;
        for (int t = 0; t < 256; ++t) {
            int g = s_gt[t], e = s_eq[t];
            s_gt[t] = ag; s_eq[t] = ae;
            ag += g; ae += e;
        }
        s_gt[256] = ag;
    }
    __syncthreads();
    int needed = Msel - s_gt[256];
    int gp = s_gt[threadIdx.x], ep = s_eq[threadIdx.x];
    for (int i = 0; i < per; ++i) {
        int node = start + i;
        unsigned u = __float_as_uint(m[node]);
        bool g = (u > T), e2 = (u == T);
        bool sel = g || (e2 && ep < needed);
        int pos = gp + (ep < needed ? ep : needed);
        if (sel) midx[pos] = node;
        inv[node] = sel ? pos : -1;
        gp += g ? 1 : 0; ep += e2 ? 1 : 0;
    }
}

// ---------------------------------------------------------------------------
// Gather h_m = h_local[midx], pos_m = pos[midx]
// ---------------------------------------------------------------------------
__global__ void gather_kernel(const float* __restrict__ hB, const float* __restrict__ pos,
                              const int* __restrict__ midx,
                              float* __restrict__ h_m, float* __restrict__ pos_m) {
    int idx = blockIdx.x * blockDim.x + threadIdx.x;
    if (idx >= MM * HID_) return;
    int p = idx / HID_, c = idx - p * HID_;
    int nidx = midx[p];
    h_m[idx] = hB[(size_t)nidx * HID_ + c];
    if (c < 3) pos_m[p * 3 + c] = pos[nidx * 3 + c];
}

// ---------------------------------------------------------------------------
// Q = hms@vq, K = hms@vk, stored f16, K-padded to 32 with zeros
// ---------------------------------------------------------------------------
__global__ void qk_kernel(const float* __restrict__ h_m,
                          const float* __restrict__ vq, const float* __restrict__ vk,
                          _Float16* __restrict__ Qh, _Float16* __restrict__ Kh) {
    int idx = blockIdx.x * blockDim.x + threadIdx.x;
    if (idx >= MM * 32) return;
    int p = idx >> 5, c = idx & 31;
    if (c < SDIM_) {
        float q = 0.f, kk = 0.f;
#pragma unroll
        for (int k2 = 0; k2 < SDIM_; ++k2) {
            float hv = h_m[(size_t)p * HID_ + k2];
            q  += hv * vq[k2 * SDIM_ + c];
            kk += hv * vk[k2 * SDIM_ + c];
        }
        Qh[idx] = (_Float16)q; Kh[idx] = (_Float16)kk;
    } else {
        Qh[idx] = (_Float16)0.f; Kh[idx] = (_Float16)0.f;
    }
}

// ---------------------------------------------------------------------------
// adjacency bitmask over selected nodes
// ---------------------------------------------------------------------------
__global__ void adj_kernel(const int* __restrict__ src, const int* __restrict__ dst, int E_,
                           const int* __restrict__ inv, unsigned* __restrict__ adj) {
    int e = blockIdx.x * blockDim.x + threadIdx.x;
    if (e >= E_) return;
    int ms = inv[src[e]], md = inv[dst[e]];
    if (ms >= 0 && md >= 0)
        atomicOr(&adj[(size_t)ms * (MM / 32) + (md >> 5)], 1u << (md & 31));
}

// ---------------------------------------------------------------------------
// S = Q K^T / 4, masked (diag | adj) -> -1e30.  One wave per 16x16 tile.
// ---------------------------------------------------------------------------
__global__ __launch_bounds__(256)
void s_wmma_kernel(const unsigned* __restrict__ Qu, const unsigned* __restrict__ Ku,
                   const unsigned* __restrict__ adj, float* __restrict__ S) {
    int wid = threadIdx.x >> 5, lane = threadIdx.x & 31;
    int tile = blockIdx.x * 8 + wid;      // (MM/16)^2 = 16384 tiles
    int rt = tile >> 7, ct = tile & 127;
    V16HU a, b;
#pragma unroll
    for (int j = 0; j < 8; ++j) {
        int k = 2 * j + 16 * (lane >> 4);
        a.u[j] = Qu[(size_t)(rt * 16 + (lane & 15)) * 16 + (k >> 1)];
    }
#pragma unroll
    for (int j = 0; j < 8; ++j) {
        int k = 2 * j + 16 * (lane >> 4);
        b.u[j] = Ku[(size_t)(ct * 16 + (lane & 15)) * 16 + (k >> 1)];
    }
    v8f acc = {};
    acc = __builtin_amdgcn_wmma_f32_16x16x32_f16(false, a.v, false, b.v, (short)0, acc, false, false);
#pragma unroll
    for (int j = 0; j < 8; ++j) {
        int r = rt * 16 + j + 8 * (lane >> 4);
        int c = ct * 16 + (lane & 15);
        float v = acc[j] * 0.25f;
        bool blk = (r == c) || ((adj[(size_t)r * (MM / 32) + (c >> 5)] >> (c & 31)) & 1u);
        S[(size_t)r * MM + c] = blk ? -1e30f : v;
    }
}

// ---------------------------------------------------------------------------
// per-row top-16 of S
// ---------------------------------------------------------------------------
__global__ __launch_bounds__(256)
void rowtopk_kernel(const float* __restrict__ S, int* __restrict__ tk) {
    int row = blockIdx.x * blockDim.x + threadIdx.x;
    if (row >= MM) return;
    float bv[KV_]; int bi[KV_];
#pragma unroll
    for (int q = 0; q < KV_; ++q) { bv[q] = -3e38f; bi[q] = 0; }
    const float* Sr = S + (size_t)row * MM;
    for (int c = 0; c < MM; ++c) {
        float v = Sr[c];
        if (v > bv[KV_ - 1]) {
            bv[KV_ - 1] = v; bi[KV_ - 1] = c;
#pragma unroll
            for (int q = KV_ - 1; q > 0; --q) {
                if (bv[q] > bv[q - 1]) {
                    float tv = bv[q]; bv[q] = bv[q - 1]; bv[q - 1] = tv;
                    int ti = bi[q]; bi[q] = bi[q - 1]; bi[q - 1] = ti;
                }
            }
        }
    }
#pragma unroll
    for (int q = 0; q < KV_; ++q) tk[(size_t)row * KV_ + q] = bi[q];
}

// ---------------------------------------------------------------------------
// Build hierarchical edge list (E mapped edges + 2*M*KV attention edges)
// ---------------------------------------------------------------------------
__global__ void build_edges_kernel(const int* __restrict__ src, const int* __restrict__ dst,
                                   const int* __restrict__ inv, const int* __restrict__ tk,
                                   int* __restrict__ srcm, int* __restrict__ dstm,
                                   float* __restrict__ fullw, float* __restrict__ validm) {
    int e = blockIdx.x * blockDim.x + threadIdx.x;
    if (e >= ME_) return;
    if (e < EE) {
        int ms = inv[src[e]], md = inv[dst[e]];
        bool vi = (ms >= 0) && (md >= 0);
        srcm[e] = vi ? ms : 0;
        dstm[e] = vi ? md : 0;
        fullw[e] = vi ? 1.f : 0.f;
        validm[e] = vi ? 1.f : 0.f;
    } else {
        int j = e - EE;
        if (j < MKV) { srcm[e] = j >> 4; dstm[e] = tk[j]; }
        else { int j2 = j - MKV; srcm[e] = tk[j2]; dstm[e] = j2 >> 4; }
        fullw[e] = 1.f;          // overwritten with wv later
        validm[e] = 1.f;
    }
}

// ---------------------------------------------------------------------------
// attention scores for the 2*M*KV virtual edges + segment max (ordered-uint)
// ---------------------------------------------------------------------------
__global__ __launch_bounds__(256)
void att_kernel(const float* __restrict__ h_m, const float* __restrict__ pos_m,
                const int* __restrict__ srcm, const int* __restrict__ dstm,
                const float* __restrict__ aw1, const float* __restrict__ ab1,
                const float* __restrict__ aw2, const float* __restrict__ ab2,
                float* __restrict__ sc, unsigned* __restrict__ segmax) {
    __shared__ float s_aw1[40 * SDIM_];
    __shared__ float s_ab1[SDIM_];
    __shared__ float s_aw2[SDIM_];
    __shared__ float s_ab2;
    for (int t = threadIdx.x; t < 40 * SDIM_; t += blockDim.x) s_aw1[t] = aw1[t];
    if (threadIdx.x < SDIM_) { s_ab1[threadIdx.x] = ab1[threadIdx.x]; s_aw2[threadIdx.x] = aw2[threadIdx.x]; }
    if (threadIdx.x == 0) s_ab2 = ab2[0];
    __syncthreads();
    int j = blockIdx.x * blockDim.x + threadIdx.x;
    if (j >= MKV2) return;
    int sv = srcm[EE + j], dv = dstm[EE + j];
    float in40[40];
#pragma unroll
    for (int k = 0; k < SDIM_; ++k) {
        in40[k]         = h_m[(size_t)sv * HID_ + k];
        in40[SDIM_ + k] = h_m[(size_t)dv * HID_ + k];
    }
    float dx = pos_m[sv * 3 + 0] - pos_m[dv * 3 + 0];
    float dy = pos_m[sv * 3 + 1] - pos_m[dv * 3 + 1];
    float dz = pos_m[sv * 3 + 2] - pos_m[dv * 3 + 2];
    float r  = sqrtf(dx * dx + dy * dy + dz * dz);
    float rc = fmaxf(r, 1e-9f);
    float x = r * 0.1f;
    float x2 = x * x, x5 = x2 * x2 * x;
    float env = 1.f - 21.f * x5 + 35.f * x5 * x - 15.f * x5 * x2;
    if (x >= 1.f) env = 0.f;
    float amp = 0.4472135954999579f * env / rc;
#pragma unroll
    for (int nb = 0; nb < NB_; ++nb)
        in40[32 + nb] = amp * sinf((float)(nb + 1) * 3.14159265358979323846f * x);
    float out = s_ab2;
    for (int o = 0; o < SDIM_; ++o) {
        float z = s_ab1[o];
#pragma unroll
        for (int a = 0; a < 40; ++a) z += in40[a] * s_aw1[a * SDIM_ + o];
        out += fmaxf(z, 0.f) * s_aw2[o];
    }
    sc[j] = out;
    atomicMax(&segmax[dv], encf(out));
}

__global__ void softmax1_kernel(const float* __restrict__ sc, const int* __restrict__ dstm,
                                const unsigned* __restrict__ segmax,
                                float* __restrict__ ebuf, float* __restrict__ segsum) {
    int j = blockIdx.x * blockDim.x + threadIdx.x;
    if (j >= MKV2) return;
    int dv = dstm[EE + j];
    float e = expf(sc[j] - decf(segmax[dv]));
    ebuf[j] = e;
    atomicAdd(&segsum[dv], e);
}

__global__ void softmax2_kernel(const float* __restrict__ ebuf, const int* __restrict__ dstm,
                                const float* __restrict__ segsum, float* __restrict__ fullw) {
    int j = blockIdx.x * blockDim.x + threadIdx.x;
    if (j >= MKV2) return;
    int dv = dstm[EE + j];
    fullw[EE + j] = ebuf[j] / (segsum[dv] + 1e-9f);
}

// ---------------------------------------------------------------------------
// h_hier = h_mu + h_m
// ---------------------------------------------------------------------------
__global__ void hhier_kernel(const float* __restrict__ h_mu, const float* __restrict__ h_m,
                             float* __restrict__ h_hier) {
    int idx = blockIdx.x * blockDim.x + threadIdx.x;
    if (idx >= MM * HID_) return;
    h_hier[idx] = h_mu[idx] + h_m[idx];
}

// ---------------------------------------------------------------------------
// h_new = (1-m)*h_local + m*h_exp  (hhier==nullptr means h_exp == h_local for selected)
// ---------------------------------------------------------------------------
__global__ void update_kernel(float* __restrict__ hA, const float* __restrict__ hB,
                              const float* __restrict__ m, const int* __restrict__ inv,
                              const float* __restrict__ hhier, int total) {
    int idx = blockIdx.x * blockDim.x + threadIdx.x;
    if (idx >= total) return;
    int n = idx / HID_, c = idx - n * HID_;
    float hl = hB[idx];
    float mm = m[n];
    int p = inv[n];
    float hexp = 0.f;
    if (p >= 0) hexp = hhier ? hhier[(size_t)p * HID_ + c] : hl;
    hA[idx] = (1.f - mm) * hl + mm * hexp;
}

// ---------------------------------------------------------------------------
// pooling + head
// ---------------------------------------------------------------------------
__global__ void pool_kernel(const float* __restrict__ hA, const int* __restrict__ batch,
                            float* __restrict__ pooled) {
    int idx = blockIdx.x * blockDim.x + threadIdx.x;
    if (idx >= NN * EMB_) return;
    int n = idx >> 5, c = idx & 31;
    atomicAdd(&pooled[batch[n] * EMB_ + c], hA[(size_t)n * HID_ + c]);
}

__global__ void final_kernel(const float* __restrict__ pooled,
                             const float* __restrict__ pw1, const float* __restrict__ pb1,
                             const float* __restrict__ pw2, const float* __restrict__ pb2,
                             float* __restrict__ out) {
    int g = threadIdx.x;
    if (g >= NG_) return;
    float o = pb2[0];
    for (int c = 0; c < EMB_; ++c) {
        float z = pb1[c];
        for (int k = 0; k < EMB_; ++k) z += pooled[g * EMB_ + k] * pw1[k * EMB_ + c];
        o += fmaxf(z, 0.f) * pw2[c];
    }
    out[g] = o;
}

// ---------------------------------------------------------------------------
// Host launch
// ---------------------------------------------------------------------------
extern "C" void kernel_launch(void* const* d_in, const int* in_sizes, int n_in,
                              void* d_out, int out_size, void* d_ws, size_t ws_size,
                              hipStream_t stream) {
    const int*   atoms = (const int*)d_in[0];
    const float* pos   = (const float*)d_in[1];
    const int*   ei    = (const int*)d_in[2];
    const int*   batch = (const int*)d_in[3];
    const float* emb   = (const float*)d_in[4];
    const float* W0    = (const float*)d_in[5];
    const float* W12   = (const float*)d_in[6];
    const float* rw1   = (const float*)d_in[7];
    const float* rb1   = (const float*)d_in[8];
    const float* rw2   = (const float*)d_in[9];
    const float* rb2   = (const float*)d_in[10];
    const float* msw1  = (const float*)d_in[11];
    const float* msb1  = (const float*)d_in[12];
    const float* msw2  = (const float*)d_in[13];
    const float* msb2  = (const float*)d_in[14];
    const float* vq    = (const float*)d_in[15];
    const float* vk    = (const float*)d_in[16];
    const float* aw1   = (const float*)d_in[17];
    const float* ab1   = (const float*)d_in[18];
    const float* aw2   = (const float*)d_in[19];
    const float* ab2   = (const float*)d_in[20];
    const float* pw1   = (const float*)d_in[21];
    const float* pb1   = (const float*)d_in[22];
    const float* pw2   = (const float*)d_in[23];
    const float* pb2   = (const float*)d_in[24];

    const int* src = ei;
    const int* dst = ei + EE;

    // ------ workspace carve-up ------
    char* base = (char*)d_ws;
    size_t off = 0;
    auto alloc = [&](size_t bytes) -> void* {
        void* r = base + off;
        off = (off + bytes + 255) & ~(size_t)255;
        return r;
    };
    float*    hA      = (float*)alloc((size_t)NN * HID_ * 4);
    float*    hB      = (float*)alloc((size_t)NN * HID_ * 4);
    float*    coeffE  = (float*)alloc((size_t)EE * SH_ * 4);
    float*    coeffM  = (float*)alloc((size_t)ME_ * SH_ * 4);
    float*    mval    = (float*)alloc((size_t)NN * 4);
    int*      inv     = (int*)alloc((size_t)NN * 4);
    int*      midx    = (int*)alloc((size_t)MM * 4);
    float*    h_m     = (float*)alloc((size_t)MM * HID_ * 4);
    float*    h_mu    = (float*)alloc((size_t)MM * HID_ * 4);
    float*    h_hier  = (float*)alloc((size_t)MM * HID_ * 4);
    float*    pos_m   = (float*)alloc((size_t)MM * 3 * 4);
    _Float16* Qh      = (_Float16*)alloc((size_t)MM * 32 * 2);
    _Float16* Kh      = (_Float16*)alloc((size_t)MM * 32 * 2);
    float*    Smat    = (float*)alloc((size_t)MM * MM * 4);
    int*      tk      = (int*)alloc((size_t)MM * KV_ * 4);
    unsigned* adj     = (unsigned*)alloc((size_t)MM * (MM / 32) * 4);
    int*      srcm    = (int*)alloc((size_t)ME_ * 4);
    int*      dstm    = (int*)alloc((size_t)ME_ * 4);
    float*    fullw   = (float*)alloc((size_t)ME_ * 4);
    float*    validm  = (float*)alloc((size_t)ME_ * 4);
    float*    scb     = (float*)alloc((size_t)MKV2 * 4);
    float*    ebuf    = (float*)alloc((size_t)MKV2 * 4);
    unsigned* segmax  = (unsigned*)alloc((size_t)MM * 4);
    float*    segsum  = (float*)alloc((size_t)MM * 4);
    unsigned* Wpack   = (unsigned*)alloc((size_t)18 * 81 * 8 * 32 * 4);
    unsigned* Tbuf    = (unsigned*)alloc(256);
    float*    pooled  = (float*)alloc((size_t)NG_ * EMB_ * 4);
    (void)ws_size; (void)n_in; (void)in_sizes; (void)out_size;

    const int NC = NN * HID_;
    const int MC = MM * HID_;
    auto cdiv = [](int a, int b) { return (a + b - 1) / b; };

    embed_kernel<<<cdiv(NC, 256), 256, 0, stream>>>(atoms, emb, hA, NC);

    for (int i = 0; i < NL_; ++i) {
        const float* W = (i == 0) ? W0 : (W12 + (size_t)(i - 1) * HID_ * SH_ * HID_);
        int Kin = i ? HID_ : EMB_;
        int nchunks = SH_ * Kin / 32;
        const float* rw1i = rw1 + i * NB_ * MLPD_;
        const float* rb1i = rb1 + i * MLPD_;
        const float* rw2i = rw2 + i * MLPD_ * SH_;
        const float* rb2i = rb2 + i * SH_;

        wpack_kernel<<<cdiv(18 * nchunks * 8 * 32, 256), 256, 0, stream>>>(W, Kin, nchunks, Wpack);
        fill_f32<<<cdiv(NC, 256), 256, 0, stream>>>(hB, NC, 0.f);
        coeff_kernel<<<cdiv(EE, 256), 256, 0, stream>>>(pos, src, dst, EE,
                                                        rw1i, rb1i, rw2i, rb2i,
                                                        nullptr, nullptr, coeffE);
        {
            dim3 g(cdiv(EE, 64), 3);
            conv_wmma_kernel<<<g, 128, 0, stream>>>(hA, Kin, src, dst, coeffE, EE,
                                                    Wpack, nchunks, hB);
        }
        int wprev = i ? HID_ : EMB_;
        addpad_kernel<<<cdiv(NN * wprev, 256), 256, 0, stream>>>(hA, hB, wprev, NN);
        mgate_kernel<<<cdiv(NN, 256), 256, 0, stream>>>(hB,
                                                        msw1 + i * SDIM_ * MSH_, msb1 + i * MSH_,
                                                        msw2 + i * MSH_, msb2 + i, mval, NN);
        thresh_kernel<<<1, 256, 0, stream>>>(mval, NN, MM, Tbuf);
        select_kernel<<<1, 256, 0, stream>>>(mval, NN, MM, Tbuf, midx, inv);

        if (i == 0) {
            update_kernel<<<cdiv(NC, 256), 256, 0, stream>>>(hA, hB, mval, inv, nullptr, NC);
            continue;
        }

        gather_kernel<<<cdiv(MC, 256), 256, 0, stream>>>(hB, pos, midx, h_m, pos_m);
        qk_kernel<<<cdiv(MM * 32, 256), 256, 0, stream>>>(h_m, vq + i * SDIM_ * SDIM_,
                                                          vk + i * SDIM_ * SDIM_, Qh, Kh);
        fill_u32<<<cdiv(MM * (MM / 32), 256), 256, 0, stream>>>(adj, MM * (MM / 32), 0u);
        adj_kernel<<<cdiv(EE, 256), 256, 0, stream>>>(src, dst, EE, inv, adj);
        s_wmma_kernel<<<(MM / 16) * (MM / 16) / 8, 256, 0, stream>>>(
            (const unsigned*)Qh, (const unsigned*)Kh, adj, Smat);
        rowtopk_kernel<<<cdiv(MM, 256), 256, 0, stream>>>(Smat, tk);
        build_edges_kernel<<<cdiv(ME_, 256), 256, 0, stream>>>(src, dst, inv, tk,
                                                               srcm, dstm, fullw, validm);
        fill_u32<<<cdiv(MM, 256), 256, 0, stream>>>(segmax, MM, 0u);
        fill_f32<<<cdiv(MM, 256), 256, 0, stream>>>(segsum, MM, 0.f);
        att_kernel<<<cdiv(MKV2, 256), 256, 0, stream>>>(h_m, pos_m, srcm, dstm,
                                                        aw1 + i * 40 * SDIM_, ab1 + i * SDIM_,
                                                        aw2 + i * SDIM_, ab2 + i, scb, segmax);
        softmax1_kernel<<<cdiv(MKV2, 256), 256, 0, stream>>>(scb, dstm, segmax, ebuf, segsum);
        softmax2_kernel<<<cdiv(MKV2, 256), 256, 0, stream>>>(ebuf, dstm, segsum, fullw);
        coeff_kernel<<<cdiv(ME_, 256), 256, 0, stream>>>(pos_m, srcm, dstm, ME_,
                                                         rw1i, rb1i, rw2i, rb2i,
                                                         fullw, validm, coeffM);
        fill_f32<<<cdiv(MC, 256), 256, 0, stream>>>(h_mu, MC, 0.f);
        {
            dim3 g(cdiv(ME_, 64), 3);
            conv_wmma_kernel<<<g, 128, 0, stream>>>(h_m, HID_, srcm, dstm, coeffM, ME_,
                                                    Wpack, nchunks, h_mu);
        }
        hhier_kernel<<<cdiv(MC, 256), 256, 0, stream>>>(h_mu, h_m, h_hier);
        update_kernel<<<cdiv(NC, 256), 256, 0, stream>>>(hA, hB, mval, inv, h_hier, NC);
    }

    fill_f32<<<1, 256, 0, stream>>>(pooled, NG_ * EMB_, 0.f);
    pool_kernel<<<cdiv(NN * EMB_, 256), 256, 0, stream>>>(hA, batch, pooled);
    final_kernel<<<1, 32, 0, stream>>>(pooled, pw1, pb1, pw2, pb2, (float*)d_out);
}
